// Shakkala_22316650070366
// MI455X (gfx1250) — matrix operand order repaired and checked by reading, not compiled, not run on hardware
//
#include <hip/hip_runtime.h>
#include <cstdint>
#include <cstddef>

// ---------------------------------------------------------------------------
// Shakkala (embed -> BiLSTM -> BN -> 2x BiLSTM -> dense -> softmax) for gfx1250.
// bf16 WMMA (v_wmma_f32_16x16x32_bf16) for all GEMMs, f32 accumulate.
// Async global->LDS staging (ASYNCcnt) for the input-projection GEMM.
// ---------------------------------------------------------------------------

typedef __attribute__((ext_vector_type(16))) __bf16 v16bf;
typedef __attribute__((ext_vector_type(8)))  float  v8f;
typedef __attribute__((ext_vector_type(4)))  int    v4i;

#define S_LEN  512
#define BATCH  64
#define BSPOS  (S_LEN * BATCH)   // 32768 positions
#define BN_EPS 0.001f

#if defined(__has_builtin)
#if __has_builtin(__builtin_amdgcn_global_load_async_to_lds_b128) && \
    __has_builtin(__builtin_amdgcn_s_wait_asynccnt)
#define USE_ASYNC_LDS 1
#endif
#endif
#ifndef USE_ASYNC_LDS
#define USE_ASYNC_LDS 0
#endif

#if USE_ASYNC_LDS
// GLOBAL_LOAD_ASYNC_TO_LDS_B128: builtin expects (int4 AS1*, int4 AS3*, imm, imm).
static __device__ __forceinline__ void async_copy16(const void* g, void* l) {
    __builtin_amdgcn_global_load_async_to_lds_b128(
        (__attribute__((address_space(1))) v4i*)g,
        (__attribute__((address_space(3))) v4i*)l, 0, 0);
}
#endif

static __device__ __forceinline__ float hsig(float v) {
    return fminf(fmaxf(0.2f * v + 0.5f, 0.0f), 1.0f);
}

// A-fragment (16x32 bf16, wave32) element -> K index (contiguous, matches B):
// lanes 0-15 : elem e holds K=e (0..15)
// lanes 16-31: elem e holds K=16+e (16..31)
static __device__ __forceinline__ int a_kidx(int lane, int e) {
    return ((lane < 16) ? 0 : 16) + e;
}
// B-fragment (32x16 bf16): lane&15 = N column; lanes<16 hold K=0..15 (elem e),
// lanes>=16 hold K=16..31 (elem e).

// ---------------------------------------------------------------------------
// f32 -> bf16 weight conversion with K padding (row-major [rows, K] -> [rows, Kp])
// ---------------------------------------------------------------------------
__global__ void cvt_wpad_k(const float* __restrict__ src, __bf16* __restrict__ dst,
                           int rows, int K, int Kp) {
    int i = blockIdx.x * blockDim.x + threadIdx.x;
    if (i >= rows * Kp) return;
    int r = i / Kp, k = i - r * Kp;
    dst[i] = (__bf16)((k < K) ? src[r * K + k] : 0.0f);
}

// plain f32 -> bf16
__global__ void cvt_bf16_k(const float* __restrict__ src, __bf16* __restrict__ dst, int n) {
    int i = blockIdx.x * blockDim.x + threadIdx.x;
    if (i < n) dst[i] = (__bf16)src[i];
}

// embedding gather -> bf16 activations
__global__ void embed_k(const int* __restrict__ x, const float* __restrict__ emb,
                        __bf16* __restrict__ X0) {
    int i = blockIdx.x * blockDim.x + threadIdx.x;
    if (i >= BSPOS * 288) return;
    int p = i / 288, j = i - p * 288;
    X0[i] = (__bf16)emb[x[p] * 288 + j];
}

// BatchNorm (inference) + f32 -> bf16
__global__ void bn_cvt_k(const float* __restrict__ src, __bf16* __restrict__ dst,
                         const float* __restrict__ gamma, const float* __restrict__ beta,
                         const float* __restrict__ mean, const float* __restrict__ var,
                         int C, int n) {
    int i = blockIdx.x * blockDim.x + threadIdx.x;
    if (i >= n) return;
    int j = i % C;
    float v = (src[i] - mean[j]) * rsqrtf(var[j] + BN_EPS) * gamma[j] + beta[j];
    dst[i] = (__bf16)v;
}

// ---------------------------------------------------------------------------
// Input projection GEMM:  XG[M,N] = X[M,K] @ W[N,K]^T + bias   (bf16 in, bf16 out)
// Block = 256 threads (8 waves), 64x64 macro tile, K stepped by 32 through LDS.
// Tiles staged with GLOBAL_LOAD_ASYNC_TO_LDS_B128 (ASYNCcnt) when available.
// ---------------------------------------------------------------------------
__global__ __launch_bounds__(256)
void gemm_xg_k(const __bf16* __restrict__ X, const __bf16* __restrict__ W,
               const float* __restrict__ bias, __bf16* __restrict__ XG,
               int N, int K) {
    __shared__ __align__(16) __bf16 At[64 * 32];
    __shared__ __align__(16) __bf16 Bt[64 * 32];
    const int tid  = threadIdx.x;
    const int lane = tid & 31;
    const int wv   = tid >> 5;
    const int mbase = blockIdx.y * 64;
    const int nbase = blockIdx.x * 64;
    const int t0 = wv * 2, t1 = wv * 2 + 1;

    v8f acc0 = {}, acc1 = {};
    const int mloc = lane & 15;              // A row within 16, also B column
    const int kof  = (lane < 16) ? 0 : 16;   // B-frag K offset
    const int rA   = tid >> 2;               // staging: row, 16B (8 elem) chunk
    const int c8   = (tid & 3) * 8;

    for (int kb = 0; kb < K; kb += 32) {
#if USE_ASYNC_LDS
        // 64x32 bf16 tile = 4KB = 256 threads x 16B; ASYNCcnt-tracked DMA to LDS.
        async_copy16(X + (size_t)(mbase + rA) * K + kb + c8, At + tid * 8);
        async_copy16(W + (size_t)(nbase + rA) * K + kb + c8, Bt + tid * 8);
        __builtin_amdgcn_s_wait_asynccnt(0);
#else
        for (int i = tid; i < 64 * 32; i += 256) {
            int r = i >> 5, c = i & 31;
            At[i] = X[(size_t)(mbase + r) * K + kb + c];
            Bt[i] = W[(size_t)(nbase + r) * K + kb + c];
        }
#endif
        __syncthreads();
        {
            int tm = t0 >> 2, tn = t0 & 3;
            v16bf a = {}, b = {};
            #pragma unroll
            for (int e = 0; e < 16; ++e) a[e] = At[(tm * 16 + mloc) * 32 + a_kidx(lane, e)];
            #pragma unroll
            for (int e = 0; e < 16; ++e) b[e] = Bt[(tn * 16 + mloc) * 32 + kof + e];
            acc0 = __builtin_amdgcn_wmma_f32_16x16x32_bf16(false, a, false, b,
                                                           (short)0, acc0, false, false);
        }
        {
            int tm = t1 >> 2, tn = t1 & 3;
            v16bf a = {}, b = {};
            #pragma unroll
            for (int e = 0; e < 16; ++e) a[e] = At[(tm * 16 + mloc) * 32 + a_kidx(lane, e)];
            #pragma unroll
            for (int e = 0; e < 16; ++e) b[e] = Bt[(tn * 16 + mloc) * 32 + kof + e];
            acc1 = __builtin_amdgcn_wmma_f32_16x16x32_bf16(false, a, false, b,
                                                           (short)0, acc1, false, false);
        }
        __syncthreads();
    }
    // epilogue: + bias, store bf16. C-frag: VGPR r -> M = r + (lane<16?0:8), N = lane&15
    const int tiles[2] = { t0, t1 };
    v8f accs[2] = { acc0, acc1 };
    #pragma unroll
    for (int q = 0; q < 2; ++q) {
        int tm = tiles[q] >> 2, tn = tiles[q] & 3;
        int nglob = nbase + tn * 16 + (lane & 15);
        float bv = bias[nglob];
        #pragma unroll
        for (int r = 0; r < 8; ++r) {
            int m = tm * 16 + r + ((lane < 16) ? 0 : 8);
            XG[(size_t)(mbase + m) * N + nglob] = (__bf16)(accs[q][r] + bv);
        }
    }
}

// ---------------------------------------------------------------------------
// Persistent recurrent kernel: one block per (batch-tile of 16, direction).
// LDS: h (bf16, WMMA A operand), activated gates (bf16). c state in VGPRs.
// Per step: G = h @ Whh^T via WMMA, +xg, activations, c/h update, barriers.
// ---------------------------------------------------------------------------
__global__ __launch_bounds__(256)
void bilstm_rec_k(const __bf16* __restrict__ XGf, const __bf16* __restrict__ XGb,
                  const __bf16* __restrict__ Wf,  const __bf16* __restrict__ Wb,
                  float* __restrict__ OUT, int H, int Hp, int G4, int OW) {
    __shared__ __align__(16) __bf16 h_bf[16 * 288];     // [16, Hp], zero-padded K
    __shared__ __align__(16) __bf16 gact[16 * 1152];    // [16, G4] activated gates

    const int tid  = threadIdx.x;
    const int lane = tid & 31;
    const int wv   = tid >> 5;
    const bool rev = (blockIdx.y == 1);
    const __bf16* XG = rev ? XGb : XGf;
    const __bf16* W  = rev ? Wb  : Wf;
    const int mbase  = blockIdx.x * 16;
    const int Ntiles = G4 >> 4;
    const int dirOff = rev ? H : 0;

    float creg[18];                        // c state: up to 16*288/256 = 18 per thread
    {
        int s = 0;
        for (int idx = tid; idx < 16 * H; idx += 256) creg[s++] = 0.0f;
    }
    for (int i = tid; i < 16 * Hp; i += 256) h_bf[i] = (__bf16)0.0f;
    __syncthreads();

    const int mloc = lane & 15;
    const int kof  = (lane < 16) ? 0 : 16;

    for (int t = 0; t < S_LEN; ++t) {
        int tt = rev ? (S_LEN - 1 - t) : t;

        // ---- GEMM phase: gates = h @ Whh^T  (+xg, activation in epilogue) ----
        for (int tau = wv; tau < Ntiles; tau += 8) {
            v8f acc = {};
            const __bf16* Wrow = W + (size_t)(tau * 16 + mloc) * Hp;
            for (int kb = 0; kb < Hp; kb += 32) {
                __builtin_prefetch(Wrow + kb + 32, 0, 0);   // global_prefetch_b8
                v16bf a = {}, b = {};
                #pragma unroll
                for (int e = 0; e < 16; ++e) a[e] = h_bf[mloc * Hp + kb + a_kidx(lane, e)];
                #pragma unroll
                for (int e = 0; e < 16; ++e) b[e] = Wrow[kb + kof + e];
                acc = __builtin_amdgcn_wmma_f32_16x16x32_bf16(false, a, false, b,
                                                              (short)0, acc, false, false);
            }
            int nglob = tau * 16 + (lane & 15);
            int gate  = nglob / H;                       // 0:i 1:f 2:g 3:o
            #pragma unroll
            for (int r = 0; r < 8; ++r) {
                int mm = r + ((lane < 16) ? 0 : 8);
                size_t row = (size_t)(mbase + mm) * S_LEN + tt;
                float v = acc[r] + (float)XG[row * G4 + nglob];
                float g = (gate == 2) ? tanhf(v) : hsig(v);
                gact[mm * G4 + nglob] = (__bf16)g;
            }
        }
        __syncthreads();

        // ---- elementwise c/h update ----
        int s = 0;
        for (int idx = tid; idx < 16 * H; idx += 256, ++s) {
            int mm = idx / H, j = idx - mm * H;
            float ig = (float)gact[mm * G4 + j];
            float fg = (float)gact[mm * G4 + H + j];
            float gg = (float)gact[mm * G4 + 2 * H + j];
            float og = (float)gact[mm * G4 + 3 * H + j];
            float c  = fg * creg[s] + ig * gg;
            creg[s]  = c;
            float h  = og * tanhf(c);
            h_bf[mm * Hp + j] = (__bf16)h;
            OUT[((size_t)(mbase + mm) * S_LEN + tt) * OW + dirOff + j] = h;
        }
        __syncthreads();
    }
}

// ---------------------------------------------------------------------------
// Dense(28) + softmax: one wave32 per position, lane = output class.
// ---------------------------------------------------------------------------
__global__ __launch_bounds__(128)
void dense_softmax_k(const float* __restrict__ Hin, const float* __restrict__ Wd,
                     const float* __restrict__ bd, float* __restrict__ out) {
    const int lane = threadIdx.x & 31;
    const int wv   = threadIdx.x >> 5;
    const size_t p = (size_t)blockIdx.x * 4 + wv;
    if (p >= BSPOS) return;

    float logit = -3.0e38f;
    if (lane < 28) {
        float sacc = bd[lane];
        const float* h = Hin + p * 192;
        const float* w = Wd + lane * 192;
        for (int k = 0; k < 192; ++k) sacc += w[k] * h[k];
        logit = sacc;
    }
    float mx = logit;
    for (int o = 16; o; o >>= 1) mx = fmaxf(mx, __shfl_xor(mx, o, 32));
    float e = (lane < 28) ? __expf(logit - mx) : 0.0f;
    float sum = e;
    for (int o = 16; o; o >>= 1) sum += __shfl_xor(sum, o, 32);
    if (lane < 28) out[p * 28 + lane] = e / sum;
}

// ---------------------------------------------------------------------------
// Host launch
// ---------------------------------------------------------------------------
extern "C" void kernel_launch(void* const* d_in, const int* in_sizes, int n_in,
                              void* d_out, int out_size, void* d_ws, size_t ws_size,
                              hipStream_t stream) {
    (void)in_sizes; (void)n_in; (void)out_size; (void)ws_size;

    const int*   x      = (const int*)d_in[0];
    const float* emb    = (const float*)d_in[1];
    const float* wih[3][2] = { { (const float*)d_in[2],  (const float*)d_in[5]  },
                               { (const float*)d_in[8],  (const float*)d_in[11] },
                               { (const float*)d_in[14], (const float*)d_in[17] } };
    const float* whh[3][2] = { { (const float*)d_in[3],  (const float*)d_in[6]  },
                               { (const float*)d_in[9],  (const float*)d_in[12] },
                               { (const float*)d_in[15], (const float*)d_in[18] } };
    const float* bia[3][2] = { { (const float*)d_in[4],  (const float*)d_in[7]  },
                               { (const float*)d_in[10], (const float*)d_in[13] },
                               { (const float*)d_in[16], (const float*)d_in[19] } };
    const float* bn_gamma = (const float*)d_in[20];
    const float* bn_beta  = (const float*)d_in[21];
    const float* bn_mean  = (const float*)d_in[22];
    const float* bn_var   = (const float*)d_in[23];
    const float* dense_w  = (const float*)d_in[24];
    const float* dense_b  = (const float*)d_in[25];
    float* outp = (float*)d_out;

    const int H_[3]   = { 288, 144, 96 };
    const int Hp_[3]  = { 288, 160, 96 };   // K padded to multiple of 32
    const int Din_[3] = { 288, 576, 288 };
    const int G4_[3]  = { 1152, 576, 384 };

    // ---- workspace layout (~310 MB) ----
    char* ws = (char*)d_ws;
    size_t off = 0;
    auto wsalloc = [&](size_t bytes) -> void* {
        void* p = ws + off;
        off += (bytes + 255) & ~(size_t)255;
        return p;
    };
    __bf16* X0  = (__bf16*)wsalloc((size_t)BSPOS * 288 * 2);
    __bf16* X1  = (__bf16*)wsalloc((size_t)BSPOS * 576 * 2);
    __bf16* X2  = (__bf16*)wsalloc((size_t)BSPOS * 288 * 2);
    __bf16* XG  = (__bf16*)wsalloc((size_t)2 * BSPOS * 1152 * 2);   // both dirs, max G4
    float*  OUT = (float*)wsalloc((size_t)BSPOS * 576 * 4);
    __bf16* wihbf[3][2];
    __bf16* whhbf[3][2];
    for (int l = 0; l < 3; ++l)
        for (int d = 0; d < 2; ++d) {
            wihbf[l][d] = (__bf16*)wsalloc((size_t)G4_[l] * Din_[l] * 2);
            whhbf[l][d] = (__bf16*)wsalloc((size_t)G4_[l] * Hp_[l] * 2);
        }

    // ---- weight conversion (bf16, K-padded for whh) ----
    for (int l = 0; l < 3; ++l)
        for (int d = 0; d < 2; ++d) {
            int n1 = G4_[l] * Din_[l];
            cvt_wpad_k<<<(n1 + 255) / 256, 256, 0, stream>>>(wih[l][d], wihbf[l][d],
                                                             G4_[l], Din_[l], Din_[l]);
            int n2 = G4_[l] * Hp_[l];
            cvt_wpad_k<<<(n2 + 255) / 256, 256, 0, stream>>>(whh[l][d], whhbf[l][d],
                                                             G4_[l], H_[l], Hp_[l]);
        }

    // ---- embedding ----
    {
        int n = BSPOS * 288;
        embed_k<<<(n + 255) / 256, 256, 0, stream>>>(x, emb, X0);
    }

    const __bf16* Xin[3] = { X0, X1, X2 };
    for (int l = 0; l < 3; ++l) {
        const int G4 = G4_[l], Din = Din_[l], H = H_[l], Hp = Hp_[l];
        __bf16* XGf = XG;
        __bf16* XGb = XG + (size_t)BSPOS * G4;
        // input projections (xg = X @ w_ih^T + b), both directions
        dim3 ggrid(G4 / 64, BSPOS / 64);
        gemm_xg_k<<<ggrid, 256, 0, stream>>>(Xin[l], wihbf[l][0], bia[l][0], XGf, G4, Din);
        gemm_xg_k<<<ggrid, 256, 0, stream>>>(Xin[l], wihbf[l][1], bia[l][1], XGb, G4, Din);
        // recurrence: 4 batch-tiles x 2 directions
        bilstm_rec_k<<<dim3(BATCH / 16, 2), 256, 0, stream>>>(XGf, XGb,
                                                              whhbf[l][0], whhbf[l][1],
                                                              OUT, H, Hp, G4, 2 * H);
        // transition to next layer input
        if (l == 0) {
            int n = BSPOS * 576;
            bn_cvt_k<<<(n + 255) / 256, 256, 0, stream>>>(OUT, X1, bn_gamma, bn_beta,
                                                          bn_mean, bn_var, 576, n);
        } else if (l == 1) {
            int n = BSPOS * 288;
            cvt_bf16_k<<<(n + 255) / 256, 256, 0, stream>>>(OUT, X2, n);
        }
    }

    // ---- dense + softmax over layer-2 output (width 192) ----
    dense_softmax_k<<<BSPOS / 4, 128, 0, stream>>>(OUT, dense_w, dense_b, outp);
}
